// RelationalMemory_56934086476585
// MI455X (gfx1250) — matrix operand order repaired and checked by reading, not compiled, not run on hardware
//
#include <hip/hip_runtime.h>
#include <hip/hip_bf16.h>
#include <math.h>

// ---------------------------------------------------------------------------
// RelationalMemory on gfx1250 (MI455X).
// One persistent workgroup per batch element (64 WGs x 512 threads = 16 waves),
// looping over T=512 steps. State (3x512 f32) lives in LDS. All GEMMs run on
// v_wmma_f32_16x16x32_bf16 with fp32 accumulation; weights are pre-transposed
// to bf16 [N][K] in workspace and stay L2-resident (5.25 MB << 192 MB L2).
// ---------------------------------------------------------------------------

#define THREADS 512
#define WAVES   16

typedef __attribute__((ext_vector_type(16))) __bf16       v16bf;
typedef __attribute__((ext_vector_type(8)))  float        v8f;
typedef __attribute__((ext_vector_type(4)))  unsigned int u32x4;

struct Frag2 { u32x4 lo, hi; };   // 32B, 16B-aligned, trivially copyable

__device__ __forceinline__ v16bf load_frag(const unsigned short* p0,
                                           const unsigned short* p1) {
  Frag2 f;
  f.lo = *(const u32x4*)p0;   // 16 bytes (8 bf16)
  f.hi = *(const u32x4*)p1;   // 16 bytes (8 bf16)
  return __builtin_bit_cast(v16bf, f);
}

__device__ __forceinline__ unsigned short f2bf(float f) {
  unsigned int u = __float_as_uint(f);
  u += 0x7fffu + ((u >> 16) & 1u);       // round-to-nearest-even
  return (unsigned short)(u >> 16);
}

__device__ __forceinline__ float sigm(float x) {
  return 1.0f / (1.0f + __expf(-x));
}

__device__ __forceinline__ v8f wmma_bf16(v16bf a, v16bf b, v8f c) {
  return __builtin_amdgcn_wmma_f32_16x16x32_bf16(false, a, false, b,
                                                 (short)0, c, false, false);
}

// GEMM: (16 x 512 bf16 in LDS) x (512 x 16*NT bf16, [N][K] in global).
// Wave w owns tiles w, w+WAVES, ... ; per-tile epilogue gets the 16x16 f32 acc.
// C/D layout: lanes 0-15 hold rows 0..7 in acc[0..7] (we only use rows 0..3).
template <int NT, typename EP>
__device__ __forceinline__ void gemm512(const unsigned short* Ash,
                                        const unsigned short* BT,
                                        int tid, EP ep) {
  const int wave = tid >> 5;
  const int lane = tid & 31;
  const int mrow = lane & 15;
  const int hi   = lane >> 4;
  const unsigned short* aRow = Ash + mrow * 512;
  for (int nt = wave; nt < NT; nt += WAVES) {
    const unsigned short* bRow = BT + (size_t)(nt * 16 + mrow) * 512;
    v8f acc0 = {0.f,0.f,0.f,0.f,0.f,0.f,0.f,0.f};
    v8f acc1 = {0.f,0.f,0.f,0.f,0.f,0.f,0.f,0.f};
#pragma unroll 2
    for (int k0 = 0; k0 < 512; k0 += 64) {
      // A k order used here (contiguous, not the ISA table order): lane<16 -> K 0..15; lane>=16 -> K 16..31 (matches B)
      v16bf a0 = load_frag(aRow + k0      + hi * 16, aRow + k0      + hi * 16 + 8);
      v16bf a1 = load_frag(aRow + k0 + 32 + hi * 16, aRow + k0 + 32 + hi * 16 + 8);
      // B k order used here (contiguous, same as A): lane<16 -> K 0..15 of column N=lane; lane>=16 -> K 16..31
      v16bf b0 = load_frag(bRow + k0      + hi * 16, bRow + k0      + hi * 16 + 8);
      v16bf b1 = load_frag(bRow + k0 + 32 + hi * 16, bRow + k0 + 32 + hi * 16 + 8);
      acc0 = wmma_bf16(a0, b0, acc0);
      acc1 = wmma_bf16(a1, b1, acc1);
    }
    ep(nt, lane, acc0 + acc1);
  }
}

// Weight prep: W (f32, K x N row-major) -> WT (bf16, N x K row-major)
__global__ void prep_weight_kernel(const float* __restrict__ W,
                                   unsigned short* __restrict__ WT,
                                   int K, int N) {
  int idx = blockIdx.x * blockDim.x + threadIdx.x;
  if (idx >= K * N) return;
  int k = idx / N;
  int n = idx % N;
  WT[(size_t)n * K + k] = f2bf(W[idx]);
}

__global__ __launch_bounds__(THREADS)
void relmem_kernel(const float* __restrict__ inputs, const float* __restrict__ memory,
                   const unsigned short* __restrict__ WqT,  const float* __restrict__ bq,
                   const unsigned short* __restrict__ WkT,  const float* __restrict__ bk,
                   const unsigned short* __restrict__ WvT,  const float* __restrict__ bv,
                   const unsigned short* __restrict__ WoT,  const float* __restrict__ bo,
                   const float* __restrict__ lq1, const float* __restrict__ lq2,
                   const float* __restrict__ lk1, const float* __restrict__ lk2,
                   const unsigned short* __restrict__ Wm1T, const float* __restrict__ bm1,
                   const unsigned short* __restrict__ Wm2T, const float* __restrict__ bm2,
                   const unsigned short* __restrict__ WwT,  const float* __restrict__ p_bw,
                   const unsigned short* __restrict__ WuT,  const float* __restrict__ p_bu,
                   float* __restrict__ out) {
  __shared__ unsigned short Ash[16 * 512];   // bf16 A operand, rows 4..15 stay zero
  __shared__ float msh[3 * 512];             // memory state m
  __shared__ float xsh[512];                 // x_t
  __shared__ float qsh[3 * 512];             // q, later reused for h1
  __shared__ float ksh[4 * 512];
  __shared__ float vsh[4 * 512];
  __shared__ float nmsh[3 * 512];            // nm, then nm2
  __shared__ float gsh[3 * 1024];            // gates
  __shared__ float psh[96];                  // softmax probs (8 subheads x 3 q x 4 k)
  __shared__ float lam_sh;

  const int b   = blockIdx.x;
  const int tid = threadIdx.x;

  for (int i = tid; i < 3 * 512; i += THREADS)  msh[i] = memory[(size_t)b * 1536 + i];
  for (int i = tid; i < 12 * 512; i += THREADS) Ash[4 * 512 + i] = 0;  // zero pad rows
  if (tid == 0) {
    float s1 = 0.f, s2 = 0.f;
    for (int i = 0; i < 64; ++i) { s1 += lq1[i] * lk1[i]; s2 += lq2[i] * lk2[i]; }
    lam_sh = __expf(s1) - __expf(s2) + 0.8f;
  }
  __syncthreads();
  const float lam = lam_sh;

  for (int t = 0; t < 512; ++t) {
    // ---- stage A = kv (rows 0..2 = m, row 3 = x); row 3 persists all step ----
    for (int i = tid; i < 512; i += THREADS) {
      float xv = inputs[((size_t)b * 512 + t) * 512 + i];
      xsh[i] = xv;
      Ash[3 * 512 + i] = f2bf(xv);
    }
    for (int i = tid; i < 3 * 512; i += THREADS) Ash[i] = f2bf(msh[i]);
    __syncthreads();

    // ---- q, k, v projections ----
    gemm512<32>(Ash, WqT, tid, [&](int nt, int lane, v8f acc) {
      if (lane < 16) {
        int c = nt * 16 + lane; float bb = bq[c];
        qsh[c] = acc[0] + bb; qsh[512 + c] = acc[1] + bb; qsh[1024 + c] = acc[2] + bb;
      }
    });
    gemm512<32>(Ash, WkT, tid, [&](int nt, int lane, v8f acc) {
      if (lane < 16) {
        int c = nt * 16 + lane; float bb = bk[c];
        ksh[c] = acc[0] + bb; ksh[512 + c] = acc[1] + bb;
        ksh[1024 + c] = acc[2] + bb; ksh[1536 + c] = acc[3] + bb;
      }
    });
    gemm512<32>(Ash, WvT, tid, [&](int nt, int lane, v8f acc) {
      if (lane < 16) {
        int c = nt * 16 + lane; float bb = bv[c];
        vsh[c] = acc[0] + bb; vsh[512 + c] = acc[1] + bb;
        vsh[1024 + c] = acc[2] + bb; vsh[1536 + c] = acc[3] + bb;
      }
    });
    __syncthreads();

    // ---- differential attention: 8 subheads x 3 q x 4 k, softmax over k ----
    if (tid < 24) {
      int sh = tid / 3, qi = tid % 3;
      float sc[4]; float mx = -1e30f;
      for (int ki = 0; ki < 4; ++ki) {
        const float* qp = qsh + qi * 512 + sh * 64;
        const float* kp = ksh + ki * 512 + sh * 64;
        float s = 0.f;
        for (int cc = 0; cc < 64; ++cc) s += qp[cc] * kp[cc];
        s *= 0.125f;                       // 1/sqrt(64)
        sc[ki] = s; mx = fmaxf(mx, s);
      }
      float sum = 0.f;
      for (int ki = 0; ki < 4; ++ki) { float e = __expf(sc[ki] - mx); sc[ki] = e; sum += e; }
      float inv = 1.f / sum;
      for (int ki = 0; ki < 4; ++ki) psh[(sh * 3 + qi) * 4 + ki] = sc[ki] * inv;
    }
    __syncthreads();

    // attn[qi][d] = sum_ki (p[2h] - lam*p[2h+1]) * v[ki][d], h = d/128 -> A rows 0..2
    for (int idx = tid; idx < 3 * 512; idx += THREADS) {
      int qi = idx >> 9, d = idx & 511, h = d >> 7;
      float a = 0.f;
      for (int ki = 0; ki < 4; ++ki) {
        float pd = psh[((2 * h) * 3 + qi) * 4 + ki] - lam * psh[((2 * h + 1) * 3 + qi) * 4 + ki];
        a += pd * vsh[ki * 512 + d];
      }
      Ash[idx] = f2bf(a);
    }
    __syncthreads();

    // ---- nm = m + attn @ Wo + bo ----
    gemm512<32>(Ash, WoT, tid, [&](int nt, int lane, v8f acc) {
      if (lane < 16) {
        int c = nt * 16 + lane; float bb = bo[c];
        nmsh[c]        = msh[c]        + acc[0] + bb;
        nmsh[512 + c]  = msh[512 + c]  + acc[1] + bb;
        nmsh[1024 + c] = msh[1024 + c] + acc[2] + bb;
      }
    });
    __syncthreads();
    for (int i = tid; i < 3 * 512; i += THREADS) Ash[i] = f2bf(nmsh[i]);
    __syncthreads();

    // ---- h1 = relu(nm @ Wm1 + bm1) ----
    gemm512<32>(Ash, Wm1T, tid, [&](int nt, int lane, v8f acc) {
      if (lane < 16) {
        int c = nt * 16 + lane; float bb = bm1[c];
        qsh[c]        = fmaxf(acc[0] + bb, 0.f);
        qsh[512 + c]  = fmaxf(acc[1] + bb, 0.f);
        qsh[1024 + c] = fmaxf(acc[2] + bb, 0.f);
      }
    });
    __syncthreads();
    for (int i = tid; i < 3 * 512; i += THREADS) Ash[i] = f2bf(qsh[i]);
    __syncthreads();

    // ---- nm2 = nm + relu(h1 @ Wm2 + bm2) ----
    gemm512<32>(Ash, Wm2T, tid, [&](int nt, int lane, v8f acc) {
      if (lane < 16) {
        int c = nt * 16 + lane; float bb = bm2[c];
        nmsh[c]        += fmaxf(acc[0] + bb, 0.f);
        nmsh[512 + c]  += fmaxf(acc[1] + bb, 0.f);
        nmsh[1024 + c] += fmaxf(acc[2] + bb, 0.f);
      }
    });
    __syncthreads();
    for (int i = tid; i < 3 * 512; i += THREADS) Ash[i] = f2bf(tanhf(msh[i]));
    __syncthreads();

    // ---- gates = x @ Ww + bw + tanh(m) @ Wu + bu  (fused dual-B GEMM, N=1024) ----
    {
      const int wave = tid >> 5, lane = tid & 31, mrow = lane & 15, hi = lane >> 4;
      const unsigned short* aRow = Ash + mrow * 512;   // rows 0..2 = tanh(m), row 3 = x
      for (int nt = wave; nt < 64; nt += WAVES) {
        const unsigned short* bU = WuT + (size_t)(nt * 16 + mrow) * 512;
        const unsigned short* bW = WwT + (size_t)(nt * 16 + mrow) * 512;
        v8f aU = {0.f,0.f,0.f,0.f,0.f,0.f,0.f,0.f};
        v8f aW = {0.f,0.f,0.f,0.f,0.f,0.f,0.f,0.f};
#pragma unroll 2
        for (int k0 = 0; k0 < 512; k0 += 32) {
          v16bf af = load_frag(aRow + k0 + hi * 16, aRow + k0 + hi * 16 + 8);
          v16bf fu = load_frag(bU   + k0 + hi * 16, bU   + k0 + hi * 16 + 8);
          v16bf fw = load_frag(bW   + k0 + hi * 16, bW   + k0 + hi * 16 + 8);
          aU = wmma_bf16(af, fu, aU);      // two independent WMMA chains interleave
          aW = wmma_bf16(af, fw, aW);
        }
        if (lane < 16) {
          int c = nt * 16 + lane;
          float w3 = aW[3] + p_bw[c];      // x @ Ww row (A row 3), broadcast over S
          float bb = p_bu[c];
          gsh[c]        = aU[0] + bb + w3;
          gsh[1024 + c] = aU[1] + bb + w3;
          gsh[2048 + c] = aU[2] + bb + w3;
        }
      }
    }
    __syncthreads();

    // ---- gated update, write output, update state ----
    for (int idx = tid; idx < 3 * 512; idx += THREADS) {
      int s = idx >> 9, d = idx & 511;
      float ig = gsh[s * 1024 + d];
      float fg = gsh[s * 1024 + 512 + d];
      float val = sigm(ig) * tanhf(nmsh[idx]) + sigm(fg) * msh[idx];
      out[((size_t)b * 512 + t) * 1536 + idx] = val;
      msh[idx] = val;
    }
    __syncthreads();
  }
}

extern "C" void kernel_launch(void* const* d_in, const int* in_sizes, int n_in,
                              void* d_out, int out_size, void* d_ws, size_t ws_size,
                              hipStream_t stream) {
  (void)in_sizes; (void)n_in; (void)out_size; (void)ws_size;

  const float* inputs = (const float*)d_in[0];
  const float* memory = (const float*)d_in[1];
  const float* Wq  = (const float*)d_in[2];  const float* bq  = (const float*)d_in[3];
  const float* Wk  = (const float*)d_in[4];  const float* bk  = (const float*)d_in[5];
  const float* Wv  = (const float*)d_in[6];  const float* bv  = (const float*)d_in[7];
  const float* Wo  = (const float*)d_in[8];  const float* bo  = (const float*)d_in[9];
  const float* lq1 = (const float*)d_in[10]; const float* lq2 = (const float*)d_in[11];
  const float* lk1 = (const float*)d_in[12]; const float* lk2 = (const float*)d_in[13];
  const float* Wm1 = (const float*)d_in[14]; const float* bm1 = (const float*)d_in[15];
  const float* Wm2 = (const float*)d_in[16]; const float* bm2 = (const float*)d_in[17];
  const float* Ww  = (const float*)d_in[18]; const float* bw  = (const float*)d_in[19];
  const float* Wu  = (const float*)d_in[20]; const float* bu  = (const float*)d_in[21];

  unsigned short* ws   = (unsigned short*)d_ws;
  unsigned short* WqT  = ws;
  unsigned short* WkT  = WqT  + 512 * 512;
  unsigned short* WvT  = WkT  + 512 * 512;
  unsigned short* WoT  = WvT  + 512 * 512;
  unsigned short* Wm1T = WoT  + 512 * 512;
  unsigned short* Wm2T = Wm1T + 512 * 512;
  unsigned short* WwT  = Wm2T + 512 * 512;
  unsigned short* WuT  = WwT  + 512 * 1024;

  const int n512  = 512 * 512;
  const int n1024 = 512 * 1024;
  prep_weight_kernel<<<(n512  + 255) / 256, 256, 0, stream>>>(Wq,  WqT,  512, 512);
  prep_weight_kernel<<<(n512  + 255) / 256, 256, 0, stream>>>(Wk,  WkT,  512, 512);
  prep_weight_kernel<<<(n512  + 255) / 256, 256, 0, stream>>>(Wv,  WvT,  512, 512);
  prep_weight_kernel<<<(n512  + 255) / 256, 256, 0, stream>>>(Wo,  WoT,  512, 512);
  prep_weight_kernel<<<(n512  + 255) / 256, 256, 0, stream>>>(Wm1, Wm1T, 512, 512);
  prep_weight_kernel<<<(n512  + 255) / 256, 256, 0, stream>>>(Wm2, Wm2T, 512, 512);
  prep_weight_kernel<<<(n1024 + 255) / 256, 256, 0, stream>>>(Ww,  WwT,  512, 1024);
  prep_weight_kernel<<<(n1024 + 255) / 256, 256, 0, stream>>>(Wu,  WuT,  512, 1024);

  relmem_kernel<<<64, THREADS, 0, stream>>>(
      inputs, memory,
      WqT, bq, WkT, bk, WvT, bv, WoT, bo,
      lq1, lq2, lk1, lk2,
      Wm1T, bm1, Wm2T, bm2,
      WwT, bw, WuT, bu,
      (float*)d_out);
}
